// DynamicGNN_5351529251149
// MI455X (gfx1250) — matrix operand order repaired and checked by reading, not compiled, not run on hardware
//
#include <hip/hip_runtime.h>
#include <hip/hip_bf16.h>

// CDNA5 / gfx1250: wave32, WMMA 16x16x32 bf16 -> f32, TDM tensor loads
typedef __attribute__((ext_vector_type(16))) __bf16       v16bf;
typedef __attribute__((ext_vector_type(8)))  float        v8f;
typedef __attribute__((ext_vector_type(4)))  unsigned int v4u;
typedef __attribute__((ext_vector_type(8)))  int          v8i;
typedef __attribute__((ext_vector_type(4)))  int          v4i;

#define GDIM 64            // feature dim D
#define NEG_SLOPE 0.2f

#if __has_builtin(__builtin_amdgcn_tensor_load_to_lds)
#define USE_TDM 1
#else
#define USE_TDM 0
#endif

__device__ __forceinline__ void wait_tensorcnt0() {
#if __has_builtin(__builtin_amdgcn_s_wait_tensorcnt)
    __builtin_amdgcn_s_wait_tensorcnt(0);
#else
    asm volatile("s_wait_tensorcnt 0x0" ::: "memory");
#endif
}

// ---------------------------------------------------------------------------
// h[n,:] = x[n,:] @ W[n,:,:]   (einsum 'nd,nde->ne'), one wave per node.
// W[n] (64x64 fp32, 16KB) is DMA'd into LDS by the Tensor Data Mover (one D#
// per node, TENSORcnt-tracked); A-fragments read the K-major (transposed)
// pattern out of LDS with on-the-fly bf16 conversion; B-fragments are the
// x-vector replicated across the (wasted) N dimension of the 16x16x32 tile.
// 2 waves / 64-thread block; LDS = 2*16KB (W fp32) + 2*256B (x).
// ---------------------------------------------------------------------------
__global__ __launch_bounds__(64) void gemv_wmma(const float* __restrict__ x,
                                                const float* __restrict__ w,
                                                float* __restrict__ h,
                                                int n_nodes)
{
    __shared__ float Wf[2][GDIM * GDIM];    // 2 x 16 KB, row-major [d_in][d_out]
    __shared__ float xl[2][GDIM];           // 2 x 256 B

    const int lane = threadIdx.x & 31;
    const int wv   = threadIdx.x >> 5;
    int node = blockIdx.x * 2 + wv;
    const bool live = (node < n_nodes);
    if (!live) node = 0;                    // keep wave fully active (EXEC all-1s for WMMA)

    // ---- stage W[n] (4096 f32 = 16 KB) -> LDS ----
#if USE_TDM
    {
        // Tensor DMA descriptor (CDNA5 ISA):
        // 2D tensor, 4-byte elements, 64x64 tile == whole tensor, contiguous.
        const unsigned lds_base = (unsigned)(size_t)(void*)&Wf[wv][0];
        const unsigned long long ga =
            (unsigned long long)(const void*)(w + (size_t)node * (GDIM * GDIM));
        v4u g0;
        g0[0] = 1u;                                    // count=1, user mode
        g0[1] = lds_base;                              // lds_addr (bytes)
        g0[2] = (unsigned)(ga & 0xFFFFFFFFu);          // global_addr[31:0]
        g0[3] = (unsigned)((ga >> 32) & 0x01FFFFFFu)   // global_addr[56:32]
              | 0x80000000u;                           // type=2 ("image") in [127:126]
        v8i g1;
        g1[0] = (int)(2u << 16);        // wg_mask=0, data_size=2 (4 bytes)
        g1[1] = (int)(64u << 16);       // tensor_dim0[15:0]=64 in bits[79:64->31:16]
        g1[2] = (int)(64u << 16);       // tensor_dim0 hi=0; tensor_dim1[15:0]=64
        g1[3] = (int)(64u << 16);       // tensor_dim1 hi=0; tile_dim0=64
        g1[4] = 64;                     // tile_dim1=64, tile_dim2=0
        g1[5] = 64;                     // tensor_dim0_stride[31:0]=64
        g1[6] = 0;                      // stride hi, tensor_dim1_stride lo
        g1[7] = 0;
        const v4i gz = {0, 0, 0, 0};
#if defined(__clang_major__) && (__clang_major__ >= 23)
        const v8i gz8 = {0, 0, 0, 0, 0, 0, 0, 0};
        __builtin_amdgcn_tensor_load_to_lds(g0, g1, gz, gz, gz8, 0);
#else
        __builtin_amdgcn_tensor_load_to_lds(g0, g1, gz, gz, 0);
#endif
    }
#else
    {
        const float4* wp = (const float4*)(w + (size_t)node * (GDIM * GDIM));
        float4* Wf4 = (float4*)&Wf[wv][0];
        #pragma unroll 4
        for (int i = lane; i < (GDIM * GDIM) / 4; i += 32) Wf4[i] = wp[i];
    }
#endif

    // ---- stage x[n] ----
    xl[wv][lane]      = x[(size_t)node * GDIM + lane];
    xl[wv][lane + 32] = x[(size_t)node * GDIM + lane + 32];

#if USE_TDM
    wait_tensorcnt0();                  // W tile resident in LDS
#endif
    __syncthreads();                    // x staging visible across lanes

    // ---- B fragments: B[k, ncol] = x[kb + k] for every column ----
    // 32x16 bf16 B layout: lanes 0-15 hold K=0..7,16..23; lanes 16-31 hold K=8..15,24..31.
    const int koff = (lane >= 16) ? 8 : 0;
    v16bf bf0, bf1;
    #pragma unroll
    for (int t = 0; t < 16; ++t) {
        bf0[t] = (__bf16)xl[wv][ 0 + koff + t + (t >= 8 ? 8 : 0)];
        bf1[t] = (__bf16)xl[wv][32 + koff + t + (t >= 8 ? 8 : 0)];
    }

    // ---- A fragments + WMMA: A[m,k] = W[kb+k][eb+m] (cvt f32->bf16 on read) ----
    const int m  = lane & 15;
    const int k2 = (lane >> 4) * 8;     // 0 or 8 (K-half select)
    v8f acc[4];
    #pragma unroll
    for (int eb = 0; eb < 4; ++eb) {
        v8f c = {};
        #pragma unroll
        for (int ch = 0; ch < 2; ++ch) {
            const int kb = ch * 32;
            v16bf a;
            #pragma unroll
            for (int t = 0; t < 8; ++t) {
                a[t]     = (__bf16)Wf[wv][(kb + k2 + t)      * GDIM + eb * 16 + m];
                a[t + 8] = (__bf16)Wf[wv][(kb + 16 + k2 + t) * GDIM + eb * 16 + m];
            }
            c = __builtin_amdgcn_wmma_f32_16x16x32_bf16(
                    false, a, false, (ch ? bf1 : bf0), (short)0, c, false, false);
        }
        acc[eb] = c;
    }

    // ---- store: D VGPR r holds M=r (lanes 0-15) / M=8+r (lanes 16-31);
    //      all 16 N-columns identical -> half-wave predicated store.
    if (((lane & 15) < 8) && live) {
        const int r = lane & 7;
        const int e = ((lane >> 4) << 3) + r;
        #pragma unroll
        for (int eb = 0; eb < 4; ++eb) {
            float v = acc[eb][0];
            #pragma unroll
            for (int rr = 1; rr < 8; ++rr) v = (r == rr) ? acc[eb][rr] : v;
            h[(size_t)node * GDIM + eb * 16 + e] = v;
        }
    }
}

// ---------------------------------------------------------------------------
// per-edge attention score + leaky-relu + segment-max (ordered-uint atomicMax)
// one wave per edge: 256-B contiguous gathers per row, shfl_xor reduction
// ---------------------------------------------------------------------------
__global__ __launch_bounds__(256) void edge_scores(const float* __restrict__ h,
                                                   const float* __restrict__ att,
                                                   const long long* __restrict__ ei,
                                                   float* __restrict__ scores,
                                                   unsigned int* __restrict__ segmax,
                                                   int E)
{
    const int lane = threadIdx.x & 31;
    const int e = blockIdx.x * 8 + (threadIdx.x >> 5);
    if (e >= E) return;
    const long long row = ei[e];
    const long long col = ei[(size_t)E + e];
    const float* hr = h + (size_t)row * GDIM;
    const float* hc = h + (size_t)col * GDIM;
    const float* ar = att + (size_t)row * (2 * GDIM);

    float s = hr[lane]      * ar[lane]
            + hr[lane + 32] * ar[lane + 32]
            + hc[lane]      * ar[GDIM + lane]
            + hc[lane + 32] * ar[GDIM + 32 + lane];
    #pragma unroll
    for (int msk = 16; msk >= 1; msk >>= 1) s += __shfl_xor(s, msk, 32);

    if (lane == 0) {
        s = (s >= 0.0f) ? s : NEG_SLOPE * s;
        scores[e] = s;
        unsigned int u = __float_as_uint(s);
        u = (u & 0x80000000u) ? ~u : (u | 0x80000000u);   // order-preserving map
        atomicMax(&segmax[col], u);
    }
}

// ---------------------------------------------------------------------------
// ex = exp(score - segmax[col]); segsum[col] += ex  (thread per edge)
// ---------------------------------------------------------------------------
__global__ __launch_bounds__(256) void edge_expsum(float* __restrict__ scores,
                                                   const long long* __restrict__ ei,
                                                   const unsigned int* __restrict__ segmax,
                                                   float* __restrict__ segsum,
                                                   int E)
{
    const int e = blockIdx.x * 256 + threadIdx.x;
    if (e >= E) return;
    const long long col = ei[(size_t)E + e];
    const unsigned int u = segmax[col];
    const float m = (u & 0x80000000u) ? __uint_as_float(u & 0x7FFFFFFFu)
                                      : __uint_as_float(~u);
    const float ex = __expf(scores[e] - m);
    scores[e] = ex;
    atomicAdd(&segsum[col], ex);
}

// ---------------------------------------------------------------------------
// acc[col,:] += h[row,:] * (ex / segsum[col])   (wave per edge, f32 atomics)
// ---------------------------------------------------------------------------
__global__ __launch_bounds__(256) void edge_scatter(const float* __restrict__ h,
                                                    const float* __restrict__ scores,
                                                    const long long* __restrict__ ei,
                                                    const float* __restrict__ segsum,
                                                    float* __restrict__ acc,
                                                    int E)
{
    const int lane = threadIdx.x & 31;
    const int e = blockIdx.x * 8 + (threadIdx.x >> 5);
    if (e >= E) return;
    const long long row = ei[e];
    const long long col = ei[(size_t)E + e];
    const float a = scores[e] / segsum[col];
    atomicAdd(&acc[(size_t)col * GDIM + lane],      h[(size_t)row * GDIM + lane]      * a);
    atomicAdd(&acc[(size_t)col * GDIM + lane + 32], h[(size_t)row * GDIM + lane + 32] * a);
}

// ---------------------------------------------------------------------------
// per-layer init: acc = bias (broadcast), segmax = ordered(-inf)=0, segsum = 0
// ---------------------------------------------------------------------------
__global__ __launch_bounds__(256) void layer_init(unsigned int* __restrict__ segmax,
                                                  float* __restrict__ segsum,
                                                  float* __restrict__ acc,
                                                  const float* __restrict__ bias,
                                                  int n_nodes)
{
    const int i = blockIdx.x * 256 + threadIdx.x;
    if (i < n_nodes * GDIM) acc[i] = bias[i & (GDIM - 1)];
    if (i < n_nodes) { segmax[i] = 0u; segsum[i] = 0.0f; }
}

__global__ __launch_bounds__(256) void relu_k(const float* __restrict__ in,
                                              float* __restrict__ out, int total)
{
    const int i = blockIdx.x * 256 + threadIdx.x;
    if (i < total) out[i] = fmaxf(in[i], 0.0f);
}

// ---------------------------------------------------------------------------
extern "C" void kernel_launch(void* const* d_in, const int* in_sizes, int n_in,
                              void* d_out, int out_size, void* d_ws, size_t ws_size,
                              hipStream_t stream)
{
    const float*     x    = (const float*)d_in[0];
    const long long* ei   = (const long long*)d_in[1];
    const float*     w0   = (const float*)d_in[2];
    const float*     att0 = (const float*)d_in[3];
    const float*     b0   = (const float*)d_in[4];
    const float*     w1   = (const float*)d_in[5];
    const float*     att1 = (const float*)d_in[6];
    const float*     b1   = (const float*)d_in[7];
    float* out = (float*)d_out;

    const int N = in_sizes[0] / GDIM;
    const int E = in_sizes[1] / 2;

    char* ws = (char*)d_ws;
    float* h      = (float*)ws;  ws += (size_t)N * GDIM * sizeof(float);
    float* xmid   = (float*)ws;  ws += (size_t)N * GDIM * sizeof(float);
    float* acc    = (float*)ws;  ws += (size_t)N * GDIM * sizeof(float);
    float* scores = (float*)ws;  ws += (size_t)E * sizeof(float);
    unsigned int* segmax = (unsigned int*)ws; ws += (size_t)N * sizeof(unsigned int);
    float* segsum = (float*)ws;  ws += (size_t)N * sizeof(float);

    const int nodeBlocks  = (N + 1) / 2;            // 2 waves (nodes) per block
    const int edgeWBlocks = (E + 7) / 8;            // 8 waves (edges) per block
    const int edgeTBlocks = (E + 255) / 256;
    const int elemBlocks  = (N * GDIM + 255) / 256;

    // ---------------- layer 0 ----------------
    layer_init  <<<elemBlocks,  256, 0, stream>>>(segmax, segsum, acc, b0, N);
    gemv_wmma   <<<nodeBlocks,   64, 0, stream>>>(x, w0, h, N);
    edge_scores <<<edgeWBlocks, 256, 0, stream>>>(h, att0, ei, scores, segmax, E);
    edge_expsum <<<edgeTBlocks, 256, 0, stream>>>(scores, ei, segmax, segsum, E);
    edge_scatter<<<edgeWBlocks, 256, 0, stream>>>(h, scores, ei, segsum, acc, E);
    relu_k      <<<elemBlocks,  256, 0, stream>>>(acc, xmid, N * GDIM);

    // ---------------- layer 1 (accumulate straight into d_out) ----------------
    layer_init  <<<elemBlocks,  256, 0, stream>>>(segmax, segsum, out, b1, N);
    gemv_wmma   <<<nodeBlocks,   64, 0, stream>>>(xmid, w1, h, N);
    edge_scores <<<edgeWBlocks, 256, 0, stream>>>(h, att1, ei, scores, segmax, E);
    edge_expsum <<<edgeTBlocks, 256, 0, stream>>>(scores, ei, segmax, segsum, E);
    edge_scatter<<<edgeWBlocks, 256, 0, stream>>>(h, scores, ei, segsum, out, E);
}
